// NoMaD_Mamba_47115791237912
// MI455X (gfx1250) — matrix operand order repaired and checked by reading, not compiled, not run on hardware
//
#include <hip/hip_runtime.h>
#include <math.h>

// ---------------- model constants ----------------
#define CTX      5
#define LTOK     7            // CTX+2
#define DMODEL   512
#define DINNER   1024
#define DSTATE   64
#define DCONV    4
#define HEADDIM  64
#define NHEADS   16
#define DINPROJ  2192         // 2*DINNER + 2*DSTATE + NHEADS
#define CONVDIM  1152         // DINNER + 2*DSTATE
#define NBLOCKS  2
#define FEAT     1280
#define BATCH    2048
#define EPSLN    1e-5f

#define NTOK     (BATCH * LTOK)   // 14336 rows in the token stream

typedef unsigned short u16;

// ---------------- CDNA5 async global->LDS copy availability ----------------
#if defined(__has_builtin)
#if __has_builtin(__builtin_amdgcn_global_load_async_to_lds_b128) && \
    __has_builtin(__builtin_amdgcn_s_wait_asynccnt)
#define HAVE_ASYNC_LDS 1
#endif
#endif
#ifndef HAVE_ASYNC_LDS
#define HAVE_ASYNC_LDS 0
#endif

#if HAVE_ASYNC_LDS
// Builtin signature (from hipcc diagnostic): param 1 is
// '__attribute__((vector_size(16))) int __device__ *' (AS1 = global),
// param 2 is the LDS-side pointer (AS3), then (int offset, int cpol).
typedef int v4i_vs __attribute__((vector_size(16)));
typedef __attribute__((address_space(1))) v4i_vs as1_v4i;
typedef __attribute__((address_space(3))) v4i_vs as3_v4i;

// global -> LDS, 16 bytes per lane, no VGPR payload (ASYNCcnt-tracked).
// LDS aperture: addr[31:0] of a generic LDS pointer == LDS byte offset.
static __device__ __forceinline__ void async_cp_b128(const u16* gsrc, u16* ldst) {
    __builtin_amdgcn_global_load_async_to_lds_b128(
        (as1_v4i*)(unsigned long long)gsrc,
        (as3_v4i*)(unsigned int)(unsigned long long)ldst,
        0, 0);
}
#endif

// ---------------- WMMA types ----------------
typedef __bf16 bf16x16 __attribute__((ext_vector_type(16)));
typedef float  f32x8   __attribute__((ext_vector_type(8)));

union Frag {
    bf16x16 v;
    uint4   q[2];
};

static __device__ __forceinline__ u16 f2bf_bits(float f) {
    unsigned u = __builtin_bit_cast(unsigned, f);
    unsigned r = (u + 0x7FFFu + ((u >> 16) & 1u)) >> 16;   // round-to-nearest-even
    return (u16)r;
}

static __device__ __forceinline__ float siluf(float x) {
    return x / (1.0f + expf(-x));
}

// =====================================================================
// One-time fp32 -> bf16 flat conversion (encoder features only)
// =====================================================================
__global__ __launch_bounds__(256)
void convert_f32_to_bf16(const float* __restrict__ in, u16* __restrict__ out, long n4) {
    long i = (long)blockIdx.x * blockDim.x + threadIdx.x;
    if (i >= n4) return;
    const float4 v = ((const float4*)in)[i];
    union { u16 s[4]; uint2 u; } p;
    p.s[0] = f2bf_bits(v.x); p.s[1] = f2bf_bits(v.y);
    p.s[2] = f2bf_bits(v.z); p.s[3] = f2bf_bits(v.w);
    ((uint2*)out)[i] = p.u;
}

// =====================================================================
// One-time weight convert+transpose: W[K,N] fp32 -> Wt[N,K] bf16
// =====================================================================
__global__ __launch_bounds__(256)
void transpose_f32_to_bf16(const float* __restrict__ W, u16* __restrict__ Wt,
                           int K, int N) {
    __shared__ u16 t[32][34];
    const int n0 = blockIdx.x * 32, k0 = blockIdx.y * 32;
    const int tid = threadIdx.x;
    const int r  = tid >> 3;         // 0..31
    const int c4 = (tid & 7) * 4;    // 0,4,..,28
    const int k = k0 + r;
    #pragma unroll
    for (int j = 0; j < 4; ++j) {
        int n = n0 + c4 + j;
        float v = (n < N) ? W[(size_t)k * N + n] : 0.f;
        t[c4 + j][r] = f2bf_bits(v);          // t[n_local][k_local]
    }
    __syncthreads();
    const int n = n0 + r;
    if (n < N) {
        #pragma unroll
        for (int j = 0; j < 4; ++j)
            Wt[(size_t)n * K + k0 + c4 + j] = t[r][c4 + j];
    }
}

// =====================================================================
// bf16-WMMA GEMM:  C[M,N] = A[M,K](bf16, row-major) @ Wt[N,K](bf16)^T
// Block tile 64x128, 4 waves, wave tile 32x64 = 2x4 WMMA accumulators.
// Double-buffered LDS staging, one barrier per K-step. When available,
// staging uses GLOBAL_LOAD_ASYNC_TO_LDS_B128 (no VGPR payload, ASYNCcnt):
//   issue async copies (nxt) -> frags+WMMA (cur) -> s_wait_asynccnt 0
//   -> barrier -> swap.  Otherwise falls back to register staging with
// amdgpu_waves_per_eu(1) to lift the register cap (avoid scratch spills).
// OOB N rows are CLAMPED (not zero-selected); garbage only feeds tiles
// whose columns are never stored (N%16==0 -> tiles never straddle N).
// Epilogue MODE:
//   0: out = acc (+bias)                        (fp32)
//   1: obs-token scatter: orow=(row%B)*L+row/B  (+bias, fp32)
//   2: goal-token scatter: orow=row*L+L-1, zeroed if mask[row]==1 (fp32)
//   3: out = gelu(acc + bias)                   (stored as bf16)
//   4: v = acc+bias; Cout(fp32) = v; Resid += v
//   5: Resid += acc
// =====================================================================
template <int MODE>
__global__ __launch_bounds__(128)
__attribute__((amdgpu_waves_per_eu(1)))
void gemm_bf16_wmma(const u16* __restrict__ A, const u16* __restrict__ Wt,
                    const float* __restrict__ bias, void* __restrict__ CoutV,
                    float* __restrict__ Resid, const int* __restrict__ mask,
                    int M, int N, int K, int ldc) {
    __shared__ u16 As[2][64][40];     // 64(m) x 32(k) bf16, pitch 40, x2 buffers
    __shared__ u16 Bs[2][128][40];    // 128(n) x 32(k) bf16, pitch 40, x2 buffers

    const int tid  = threadIdx.x;
    const int lane = tid & 31;
    const int wid  = tid >> 5;
    const int wm   = (wid >> 1) * 32;
    const int wn   = (wid & 1) * 64;
    const int blockM = blockIdx.y * 64;
    const int blockN = blockIdx.x * 128;

    // per-thread staging coordinates + hoisted global pointers
    int aRow[2], aCol[2], wRow[4], wCol[4];
    const u16* aPtr[2];
    const u16* wPtr[4];
    #pragma unroll
    for (int i = 0; i < 2; ++i) {
        int idx = i * 128 + tid;
        aRow[i] = idx >> 2;
        aCol[i] = (idx & 3) * 8;
        aPtr[i] = A + (size_t)(blockM + aRow[i]) * K + aCol[i];
    }
    #pragma unroll
    for (int j = 0; j < 4; ++j) {
        int idx = j * 128 + tid;
        wRow[j] = idx >> 2;
        wCol[j] = (idx & 3) * 8;
        int gn  = blockN + wRow[j];
        if (gn > N - 1) gn = N - 1;            // clamp: keeps fast global path
        wPtr[j] = Wt + (size_t)gn * K + wCol[j];
    }

    f32x8 acc[2][4] = {};

#if HAVE_ASYNC_LDS
    // ---- prologue: async-stage tile 0 into buffer 0 ----
    #pragma unroll
    for (int i = 0; i < 2; ++i) async_cp_b128(aPtr[i], &As[0][aRow[i]][aCol[i]]);
    #pragma unroll
    for (int j = 0; j < 4; ++j) async_cp_b128(wPtr[j], &Bs[0][wRow[j]][wCol[j]]);
    __builtin_amdgcn_s_wait_asynccnt(0);
    __syncthreads();
#else
    uint4 aReg[2], wReg[4];
    #pragma unroll
    for (int i = 0; i < 2; ++i) aReg[i] = *(const uint4*)(aPtr[i]);
    #pragma unroll
    for (int j = 0; j < 4; ++j) wReg[j] = *(const uint4*)(wPtr[j]);
    #pragma unroll
    for (int i = 0; i < 2; ++i) *(uint4*)&As[0][aRow[i]][aCol[i]] = aReg[i];
    #pragma unroll
    for (int j = 0; j < 4; ++j) *(uint4*)&Bs[0][wRow[j]][wCol[j]] = wReg[j];
    __syncthreads();
#endif

    int cur = 0;
    for (int kt = 0; kt < K; kt += 32) {
        const bool hasNext = (kt + 32) < K;
        const int  nxt = cur ^ 1;

        if (hasNext) {
#if HAVE_ASYNC_LDS
            // async copies straight into the other LDS buffer: no VGPR
            // payload, latency hidden behind the WMMA section below.
            #pragma unroll
            for (int i = 0; i < 2; ++i)
                async_cp_b128(aPtr[i] + kt + 32, &As[nxt][aRow[i]][aCol[i]]);
            #pragma unroll
            for (int j = 0; j < 4; ++j)
                async_cp_b128(wPtr[j] + kt + 32, &Bs[nxt][wRow[j]][wCol[j]]);
#else
            #pragma unroll
            for (int i = 0; i < 2; ++i) aReg[i] = *(const uint4*)(aPtr[i] + kt + 32);
            #pragma unroll
            for (int j = 0; j < 4; ++j) wReg[j] = *(const uint4*)(wPtr[j] + kt + 32);
#endif
            if (tid < 64)   // advisory prefetch one more tile ahead
                __builtin_prefetch(aPtr[0] + kt + 64, 0, 1);
        }

        // ---- fragments from current buffer (CDNA5 16-bit WMMA layouts) ----
        Frag fa[2], fb[4];
        {
            const int klo = (lane >> 4) * 8;
            #pragma unroll
            for (int mi = 0; mi < 2; ++mi) {
                int r = wm + mi * 16 + (lane & 15);
                fa[mi].q[0] = *(const uint4*)&As[cur][r][klo];
                fa[mi].q[1] = *(const uint4*)&As[cur][r][klo + 16];
            }
            const int klo2 = (lane >> 4) * 8;
            #pragma unroll
            for (int ni = 0; ni < 4; ++ni) {
                int n = wn + ni * 16 + (lane & 15);
                fb[ni].q[0] = *(const uint4*)&Bs[cur][n][klo2];
                fb[ni].q[1] = *(const uint4*)&Bs[cur][n][klo2 + 16];
            }
        }
        #pragma unroll
        for (int mi = 0; mi < 2; ++mi)
            #pragma unroll
            for (int ni = 0; ni < 4; ++ni)
                acc[mi][ni] = __builtin_amdgcn_wmma_f32_16x16x32_bf16(
                    false, fa[mi].v, false, fb[ni].v, (short)0, acc[mi][ni],
                    false, false);

        if (hasNext) {
#if HAVE_ASYNC_LDS
            __builtin_amdgcn_s_wait_asynccnt(0);   // own copies landed in LDS
            __syncthreads();                        // cross-wave visibility
#else
            #pragma unroll
            for (int i = 0; i < 2; ++i) *(uint4*)&As[nxt][aRow[i]][aCol[i]] = aReg[i];
            #pragma unroll
            for (int j = 0; j < 4; ++j) *(uint4*)&Bs[nxt][wRow[j]][wCol[j]] = wReg[j];
            __syncthreads();
#endif
            cur = nxt;
        }
    }

    // ---- epilogue ----
    float* Cf = (float*)CoutV;
    u16*   Cb = (u16*)CoutV;
    const int colBase = lane & 15;
    const int rowHalf = (lane >> 4) * 8;
    #pragma unroll
    for (int mi = 0; mi < 2; ++mi) {
        #pragma unroll
        for (int ni = 0; ni < 4; ++ni) {
            int gcol = blockN + wn + ni * 16 + colBase;
            if (gcol >= N) continue;          // tiles never straddle N (N%16==0)
            #pragma unroll
            for (int r = 0; r < 8; ++r) {
                int grow = blockM + wm + mi * 16 + rowHalf + r;
                float v  = acc[mi][ni][r];
                if (MODE == 0) {
                    if (bias) v += bias[gcol];
                    Cf[(size_t)grow * ldc + gcol] = v;
                } else if (MODE == 1) {
                    v += bias[gcol];
                    int orow = (grow % BATCH) * LTOK + (grow / BATCH);
                    Cf[(size_t)orow * ldc + gcol] = v;
                } else if (MODE == 2) {
                    v += bias[gcol];
                    float keep = (mask[grow] != 1) ? 1.0f : 0.0f;
                    Cf[((size_t)grow * LTOK + (LTOK - 1)) * ldc + gcol] = v * keep;
                } else if (MODE == 3) {
                    v += bias[gcol];
                    v = 0.5f * v * (1.0f + erff(v * 0.70710678118654752f));
                    Cb[(size_t)grow * ldc + gcol] = f2bf_bits(v);
                } else if (MODE == 4) {
                    if (bias) v += bias[gcol];
                    Cf[(size_t)grow * ldc + gcol] = v;
                    Resid[(size_t)grow * ldc + gcol] += v;
                } else { // 5
                    Resid[(size_t)grow * ldc + gcol] += v;
                }
            }
        }
    }
}

// =====================================================================
// LayerNorm over rows of 512, fp32 in -> bf16 out (feeds WMMA GEMMs)
// =====================================================================
__global__ __launch_bounds__(128)
void ln_kernel(const float* __restrict__ in, const float* __restrict__ w,
               const float* __restrict__ b, u16* __restrict__ out) {
    const int row = blockIdx.x;
    const int tid = threadIdx.x;
    const float* rp = in + (size_t)row * DMODEL;
    float v[4], s = 0.f, s2 = 0.f;
    #pragma unroll
    for (int j = 0; j < 4; ++j) {
        v[j] = rp[j * 128 + tid];
        s += v[j];
        s2 += v[j] * v[j];
    }
    __shared__ float rs[128], rq[128];
    rs[tid] = s; rq[tid] = s2;
    __syncthreads();
    for (int o = 64; o > 0; o >>= 1) {
        if (tid < o) { rs[tid] += rs[tid + o]; rq[tid] += rq[tid + o]; }
        __syncthreads();
    }
    const float mean = rs[0] * (1.0f / DMODEL);
    const float var  = rq[0] * (1.0f / DMODEL) - mean * mean;
    const float inv  = rsqrtf(var + EPSLN);
    #pragma unroll
    for (int j = 0; j < 4; ++j) {
        int c = j * 128 + tid;
        out[(size_t)row * DMODEL + c] = f2bf_bits((v[j] - mean) * inv * w[c] + b[c]);
    }
}

// =====================================================================
// Causal depthwise conv1d (d_conv=4, left pad 3) + bias + SiLU
// =====================================================================
__global__ __launch_bounds__(256)
void conv_silu_kernel(const float* __restrict__ zx, const float* __restrict__ cw,
                      const float* __restrict__ cb, float* __restrict__ xbc) {
    int idx = blockIdx.x * blockDim.x + threadIdx.x;
    if (idx >= BATCH * CONVDIM) return;
    const int c = idx % CONVDIM;
    const int b = idx / CONVDIM;
    const float w0 = cw[c * DCONV + 0], w1 = cw[c * DCONV + 1];
    const float w2 = cw[c * DCONV + 2], w3 = cw[c * DCONV + 3];
    const float bb = cb[c];
    float h0 = 0.f, h1 = 0.f, h2 = 0.f;
    #pragma unroll
    for (int l = 0; l < LTOK; ++l) {
        const size_t rb = (size_t)(b * LTOK + l);
        const float xv = zx[rb * DINPROJ + DINNER + c];
        float a = w0 * h0 + w1 * h1 + w2 * h2 + w3 * xv + bb;
        h0 = h1; h1 = h2; h2 = xv;
        xbc[rb * CONVDIM + c] = siluf(a);
    }
}

// =====================================================================
// Selective-state scan. Grid (BATCH, 4); block 256 = 4 heads x 64 p.
// 64-float per-thread state held in registers.
// =====================================================================
__global__ __launch_bounds__(256)
__attribute__((amdgpu_waves_per_eu(1)))
void ssm_scan_kernel(const float* __restrict__ zx, const float* __restrict__ xbc,
                     const float* __restrict__ dtb, const float* __restrict__ Alog,
                     const float* __restrict__ Dv, float* __restrict__ y) {
    const int b  = blockIdx.x;
    const int hg = blockIdx.y;
    __shared__ float Bsh[LTOK][DSTATE], Csh[LTOK][DSTATE];
    __shared__ float dts[LTOK][NHEADS], dAs[LTOK][NHEADS];
    const int tid = threadIdx.x;

    for (int idx = tid; idx < LTOK * DSTATE; idx += 256) {
        int l = idx >> 6, n = idx & 63;
        const size_t rb = (size_t)(b * LTOK + l);
        Bsh[l][n] = xbc[rb * CONVDIM + DINNER + n];
        Csh[l][n] = xbc[rb * CONVDIM + DINNER + DSTATE + n];
    }
    for (int idx = tid; idx < LTOK * NHEADS; idx += 256) {
        int l = idx >> 4, h = idx & 15;
        const size_t rb = (size_t)(b * LTOK + l);
        float raw = zx[rb * DINPROJ + 2 * DINNER + 2 * DSTATE + h] + dtb[h];
        float dt  = (raw > 20.0f) ? raw : log1pf(expf(raw));   // softplus
        dts[l][h] = dt;
        dAs[l][h] = expf(-dt * expf(Alog[h]));
    }
    __syncthreads();

    const int h = hg * 4 + (tid >> 6);
    const int p = tid & 63;
    float st[DSTATE];
    #pragma unroll
    for (int n = 0; n < DSTATE; ++n) st[n] = 0.f;
    const float Dh = Dv[h];

    #pragma unroll
    for (int l = 0; l < LTOK; ++l) {
        const size_t rb = (size_t)(b * LTOK + l);
        const float xv   = xbc[rb * CONVDIM + h * HEADDIM + p];
        const float coef = dts[l][h] * xv;
        const float dal  = dAs[l][h];
        float accv = 0.f;
        #pragma unroll
        for (int n = 0; n < DSTATE; ++n) {
            st[n] = st[n] * dal + coef * Bsh[l][n];
            accv += st[n] * Csh[l][n];
        }
        y[rb * DINNER + h * HEADDIM + p] = accv + Dh * xv;
    }
}

// =====================================================================
// Gated RMSNorm, fp32 y/z in -> bf16 out (feeds out_proj WMMA GEMM)
// =====================================================================
__global__ __launch_bounds__(256)
void gated_rmsnorm_kernel(const float* __restrict__ zx, const float* __restrict__ rw,
                          const float* __restrict__ y, u16* __restrict__ yout) {
    const int row = blockIdx.x;
    const int tid = threadIdx.x;
    float g[4], s2 = 0.f;
    #pragma unroll
    for (int j = 0; j < 4; ++j) {
        int c = j * 256 + tid;
        float z  = zx[(size_t)row * DINPROJ + c];
        float gz = y[(size_t)row * DINNER + c] * siluf(z);
        g[j] = gz;
        s2 += gz * gz;
    }
    __shared__ float rs[256];
    rs[tid] = s2;
    __syncthreads();
    for (int o = 128; o > 0; o >>= 1) {
        if (tid < o) rs[tid] += rs[tid + o];
        __syncthreads();
    }
    const float inv = rsqrtf(rs[0] * (1.0f / DINNER) + EPSLN);
    #pragma unroll
    for (int j = 0; j < 4; ++j) {
        int c = j * 256 + tid;
        yout[(size_t)row * DINNER + c] = f2bf_bits(g[j] * inv * rw[c]);
    }
}

// =====================================================================
// Masked average pool over L tokens -> out[B, 512]
// =====================================================================
__global__ __launch_bounds__(256)
void pool_kernel(const float* __restrict__ hs, const int* __restrict__ mask,
                 float* __restrict__ out) {
    int idx = blockIdx.x * blockDim.x + threadIdx.x;
    if (idx >= BATCH * DMODEL) return;
    const int b = idx >> 9, d = idx & 511;
    const int m = mask[b];
    const float wc = m ? ((float)LTOK / (LTOK - 1)) : 1.0f;
    const float wg = m ? 0.0f : 1.0f;
    float s = 0.f;
    #pragma unroll
    for (int l = 0; l < LTOK - 1; ++l)
        s += wc * hs[((size_t)(b * LTOK + l)) * DMODEL + d];
    s += wg * hs[((size_t)(b * LTOK + LTOK - 1)) * DMODEL + d];
    out[idx] = s * (1.0f / LTOK);
}

// =====================================================================
// host orchestration
// =====================================================================
extern "C" void kernel_launch(void* const* d_in, const int* in_sizes, int n_in,
                              void* d_out, int out_size, void* d_ws, size_t ws_size,
                              hipStream_t stream) {
    const float* obs_feats = (const float*)d_in[0];
    const float* goal_feats= (const float*)d_in[1];
    const int*   gmask     = (const int*)  d_in[2];
    const float* w_obs     = (const float*)d_in[3];
    const float* b_obs     = (const float*)d_in[4];
    const float* w_goal    = (const float*)d_in[5];
    const float* b_goal    = (const float*)d_in[6];
    const float* ln1_w     = (const float*)d_in[7];
    const float* ln1_b     = (const float*)d_in[8];
    const float* ln2_w     = (const float*)d_in[9];
    const float* ln2_b     = (const float*)d_in[10];
    const float* in_proj_w = (const float*)d_in[11];
    const float* conv_w    = (const float*)d_in[12];
    const float* conv_b    = (const float*)d_in[13];
    const float* dt_bias   = (const float*)d_in[14];
    const float* A_log     = (const float*)d_in[15];
    const float* Dvec      = (const float*)d_in[16];
    const float* rms_w     = (const float*)d_in[17];
    const float* out_proj_w= (const float*)d_in[18];
    const float* mlp_w1    = (const float*)d_in[19];
    const float* mlp_b1    = (const float*)d_in[20];
    const float* mlp_w2    = (const float*)d_in[21];
    const float* mlp_b2    = (const float*)d_in[22];
    float* out = (float*)d_out;

    // -------- workspace partition --------
    float* fp = (float*)d_ws;
    float* resid = fp;                       fp += (size_t)NTOK * DMODEL;
    float* zx    = fp;                       fp += (size_t)NTOK * DINPROJ;
    float* xbc   = fp;                       fp += (size_t)NTOK * CONVDIM;
    float* yb    = fp;                       fp += (size_t)NTOK * DINNER;
    float* hsb   = fp;                       fp += (size_t)NTOK * DMODEL;
    u16* bp = (u16*)fp;
    u16* x_bf   = bp;                        bp += (size_t)NTOK * DMODEL;
    u16* ybb    = bp;                        bp += (size_t)NTOK * DINNER;
    u16* m1b    = bp;                        bp += (size_t)NTOK * 4 * DMODEL;
    u16* obs_bf = bp;                        bp += (size_t)(CTX + 1) * BATCH * FEAT;
    u16* goal_bf= bp;                        bp += (size_t)BATCH * FEAT;
    u16* wt_obs = bp;                        bp += (size_t)DMODEL * FEAT;
    u16* wt_goal= bp;                        bp += (size_t)DMODEL * FEAT;
    u16* wt_inp = bp;                        bp += (size_t)NBLOCKS * DINPROJ * DMODEL;
    u16* wt_outp= bp;                        bp += (size_t)NBLOCKS * DMODEL * DINNER;
    u16* wt_m1  = bp;                        bp += (size_t)NBLOCKS * 4 * DMODEL * DMODEL;
    u16* wt_m2  = bp;                        bp += (size_t)NBLOCKS * DMODEL * 4 * DMODEL;

    // -------- one-time bf16 conversion / weight transposition --------
    {
        long n4 = (long)(CTX + 1) * BATCH * FEAT / 4;
        convert_f32_to_bf16<<<(unsigned)((n4 + 255) / 256), 256, 0, stream>>>(obs_feats, obs_bf, n4);
        long g4 = (long)BATCH * FEAT / 4;
        convert_f32_to_bf16<<<(unsigned)((g4 + 255) / 256), 256, 0, stream>>>(goal_feats, goal_bf, g4);
    }
    transpose_f32_to_bf16<<<dim3((DMODEL + 31) / 32, FEAT / 32), 256, 0, stream>>>(w_obs,  wt_obs,  FEAT, DMODEL);
    transpose_f32_to_bf16<<<dim3((DMODEL + 31) / 32, FEAT / 32), 256, 0, stream>>>(w_goal, wt_goal, FEAT, DMODEL);
    for (int i = 0; i < NBLOCKS; ++i) {
        transpose_f32_to_bf16<<<dim3((DINPROJ + 31) / 32, DMODEL / 32), 256, 0, stream>>>(
            in_proj_w + (size_t)i * DMODEL * DINPROJ, wt_inp + (size_t)i * DINPROJ * DMODEL, DMODEL, DINPROJ);
        transpose_f32_to_bf16<<<dim3((DMODEL + 31) / 32, DINNER / 32), 256, 0, stream>>>(
            out_proj_w + (size_t)i * DINNER * DMODEL, wt_outp + (size_t)i * DMODEL * DINNER, DINNER, DMODEL);
        transpose_f32_to_bf16<<<dim3((4 * DMODEL + 31) / 32, DMODEL / 32), 256, 0, stream>>>(
            mlp_w1 + (size_t)i * DMODEL * 4 * DMODEL, wt_m1 + (size_t)i * 4 * DMODEL * DMODEL, DMODEL, 4 * DMODEL);
        transpose_f32_to_bf16<<<dim3((DMODEL + 31) / 32, (4 * DMODEL) / 32), 256, 0, stream>>>(
            mlp_w2 + (size_t)i * 4 * DMODEL * DMODEL, wt_m2 + (size_t)i * DMODEL * 4 * DMODEL, 4 * DMODEL, DMODEL);
    }

    // -------- encoders: write tokens straight into residual stream --------
    {
        dim3 g((DMODEL + 127) / 128, ((CTX + 1) * BATCH) / 64);
        gemm_bf16_wmma<1><<<g, 128, 0, stream>>>(obs_bf, wt_obs, b_obs, resid,
                                                 nullptr, nullptr,
                                                 (CTX + 1) * BATCH, DMODEL, FEAT, DMODEL);
    }
    {
        dim3 g((DMODEL + 127) / 128, BATCH / 64);
        gemm_bf16_wmma<2><<<g, 128, 0, stream>>>(goal_bf, wt_goal, b_goal, resid,
                                                 nullptr, gmask,
                                                 BATCH, DMODEL, FEAT, DMODEL);
    }

    for (int i = 0; i < NBLOCKS; ++i) {
        ln_kernel<<<NTOK, 128, 0, stream>>>(resid, ln1_w + i * DMODEL, ln1_b + i * DMODEL, x_bf);
        {
            dim3 g((DINPROJ + 127) / 128, NTOK / 64);
            gemm_bf16_wmma<0><<<g, 128, 0, stream>>>(x_bf, wt_inp + (size_t)i * DINPROJ * DMODEL,
                                                     nullptr, zx, nullptr, nullptr,
                                                     NTOK, DINPROJ, DMODEL, DINPROJ);
        }
        conv_silu_kernel<<<(BATCH * CONVDIM + 255) / 256, 256, 0, stream>>>(
            zx, conv_w + (size_t)i * CONVDIM * DCONV, conv_b + i * CONVDIM, xbc);
        {
            dim3 g(BATCH, NHEADS / 4);
            ssm_scan_kernel<<<g, 256, 0, stream>>>(zx, xbc, dt_bias + i * NHEADS,
                                                   A_log + i * NHEADS, Dvec + i * NHEADS, yb);
        }
        gated_rmsnorm_kernel<<<NTOK, 256, 0, stream>>>(zx, rms_w + i * DINNER, yb, ybb);
        {
            dim3 g((DMODEL + 127) / 128, NTOK / 64);
            gemm_bf16_wmma<5><<<g, 128, 0, stream>>>(ybb, wt_outp + (size_t)i * DMODEL * DINNER,
                                                     nullptr, nullptr, resid, nullptr,
                                                     NTOK, DMODEL, DINNER, DMODEL);
        }
        ln_kernel<<<NTOK, 128, 0, stream>>>(resid, ln2_w + i * DMODEL, ln2_b + i * DMODEL, x_bf);
        {
            dim3 g((4 * DMODEL + 127) / 128, NTOK / 64);
            gemm_bf16_wmma<3><<<g, 128, 0, stream>>>(x_bf, wt_m1 + (size_t)i * 4 * DMODEL * DMODEL,
                                                     mlp_b1 + i * 4 * DMODEL, m1b, nullptr, nullptr,
                                                     NTOK, 4 * DMODEL, DMODEL, 4 * DMODEL);
        }
        {
            dim3 g((DMODEL + 127) / 128, NTOK / 64);
            gemm_bf16_wmma<4><<<g, 128, 0, stream>>>(m1b, wt_m2 + (size_t)i * DMODEL * 4 * DMODEL,
                                                     mlp_b2 + i * DMODEL, hsb, resid, nullptr,
                                                     NTOK, DMODEL, 4 * DMODEL, DMODEL);
        }
    }

    pool_kernel<<<(BATCH * DMODEL + 255) / 256, 256, 0, stream>>>(hsb, gmask, out);
}
